// PlainAttnLayer_68848325755396
// MI455X (gfx1250) — matrix operand order repaired and checked by reading, not compiled, not run on hardware
//
#include <hip/hip_runtime.h>
#include <math.h>

typedef __attribute__((ext_vector_type(16))) _Float16 v16h;
typedef __attribute__((ext_vector_type(8)))  _Float16 v8h;
typedef __attribute__((ext_vector_type(4)))  _Float16 v4h;
typedef __attribute__((ext_vector_type(8)))  float    v8f;
typedef __attribute__((ext_vector_type(2)))  __fp16   v2fp;
typedef __attribute__((ext_vector_type(4)))  __fp16   v4fp;

#define TILE_M 128
#define TILE_N 128
#define TILE_K 32
#define GTHREADS 256

// Pack 4 floats -> 4 f16 using v_cvt_pkrtz_f16_f32 (2 packs -> 1 VGPR pair).
__device__ __forceinline__ v4h pack4(float a, float b, float c, float d) {
    v2fp lo = __builtin_amdgcn_cvt_pkrtz(a, b);
    v2fp hi = __builtin_amdgcn_cvt_pkrtz(c, d);
    v4fp r  = __builtin_shufflevector(lo, hi, 0, 1, 2, 3);
    return __builtin_bit_cast(v4h, r);
}

// Build a 16x32 f16 A-fragment for one lane from an LDS row (row-major [M][TILE_K]).
// Layout (matches B fragment): lanes 0-15 hold K=[0..16), lanes 16-31 hold
// K=[16..32); within-lane VGPR order is ascending K.
__device__ __forceinline__ v16h load_a_frag(const _Float16* __restrict__ row, int lane) {
    const int klo = (lane >> 4) * 16;
    v8h lo = *(const v8h*)(row + klo);
    v8h hi = *(const v8h*)(row + klo + 8);
    return __builtin_shufflevector(lo, hi, 0,1,2,3,4,5,6,7,8,9,10,11,12,13,14,15);
}

// Stage one K-step of A and B tiles from global (fp32) into packed f16 registers.
// BLAYOUT: 0 = B row-major [K,N]; 1 = B^T row-major [N,K].
template<int BLAYOUT>
__device__ __forceinline__ void load_tile_regs(
    const float* __restrict__ A, const float* __restrict__ Bp,
    int K, int N, int tile_m, int tile_n, int k0, int tid,
    v4h (&sa)[4], v4h (&sb)[4])
{
    #pragma unroll
    for (int it = 0; it < 4; ++it) {
        int idx  = tid + it * GTHREADS;            // 0..1023
        int rowA = idx >> 3;                       // 8 float4 per 32-float row
        int c4   = idx & 7;
        float4 v = *(const float4*)(A + (long long)(tile_m + rowA) * K + k0 + c4 * 4);
        sa[it] = pack4(v.x, v.y, v.z, v.w);
    }
    if (BLAYOUT == 0) {
        // Transpose-read: thread grabs a 4-deep K-column of one N-column.
        // Per fixed kg, consecutive threads read consecutive columns -> coalesced b32.
        #pragma unroll
        for (int it = 0; it < 4; ++it) {
            int idx = tid + it * GTHREADS;
            int col = idx & 127;
            int kg  = idx >> 7;                    // 0..7 (covers k = 4*kg..4*kg+3)
            const float* p = Bp + (long long)(k0 + kg * 4) * N + tile_n + col;
            const long long ldN = (long long)N;
            sb[it] = pack4(p[0], p[ldN], p[2 * ldN], p[3 * ldN]);
        }
    } else {
        #pragma unroll
        for (int it = 0; it < 4; ++it) {
            int idx  = tid + it * GTHREADS;
            int rowN = idx >> 3;
            int c4   = idx & 7;
            float4 v = *(const float4*)(Bp + (long long)(tile_n + rowN) * K + k0 + c4 * 4);
            sb[it] = pack4(v.x, v.y, v.z, v.w);
        }
    }
}

// Store staged registers into an LDS buffer as 64-bit writes.
template<int BLAYOUT>
__device__ __forceinline__ void store_tile_lds(
    _Float16 (* __restrict__ la)[TILE_K], _Float16 (* __restrict__ lb)[TILE_K],
    int tid, const v4h (&sa)[4], const v4h (&sb)[4])
{
    #pragma unroll
    for (int it = 0; it < 4; ++it) {
        int idx  = tid + it * GTHREADS;
        int rowA = idx >> 3;
        int c4   = idx & 7;
        *(v4h*)&la[rowA][c4 * 4] = sa[it];
    }
    #pragma unroll
    for (int it = 0; it < 4; ++it) {
        int idx = tid + it * GTHREADS;
        if (BLAYOUT == 0) {
            int col = idx & 127;
            int kg  = idx >> 7;
            *(v4h*)&lb[col][kg * 4] = sb[it];
        } else {
            int rowN = idx >> 3;
            int c4   = idx & 7;
            *(v4h*)&lb[rowN][c4 * 4] = sb[it];
        }
    }
}

// Epilogue modes:
// 0: out = acc + bias[n]
// 1: out = acc*scale + mask[b][n][m] * (-1e9)     (scores)
// 2: out = gelu(acc + bias[n])
// 3: out = acc
template<int BLAYOUT, int EPI>
__global__ __launch_bounds__(GTHREADS)
void wmma_gemm(const float* __restrict__ Ag, long long sA,
               const float* __restrict__ Bg, long long sB,
               float* __restrict__ Cg, long long sC,
               const float* __restrict__ bias,
               const int* __restrict__ maskp, long long sMask, int ldm,
               int M, int N, int K, float scale)
{
    __shared__ alignas(32) _Float16 lds_a [2][TILE_M][TILE_K];   // [buf][m][k]
    __shared__ alignas(32) _Float16 lds_bt[2][TILE_N][TILE_K];   // [buf][n][k]

    const int b = blockIdx.z;
    const float* __restrict__ A  = Ag + (long long)b * sA;
    const float* __restrict__ Bp = Bg + (long long)b * sB;
    float* __restrict__ C = Cg + (long long)b * sC;

    const int tile_n = blockIdx.x * TILE_N;
    const int tile_m = blockIdx.y * TILE_M;

    const int tid  = threadIdx.x;
    const int lane = tid & 31;
    const int wave = tid >> 5;     // 0..7
    const int wm   = wave >> 2;    // 0..1 : 64 rows each
    const int wn   = wave & 3;     // 0..3 : 32 cols each
    const int lr   = lane & 15;
    const int kh   = (lane >> 4) * 16;

    v8f acc[4][2] = {};
    v4h sa[4], sb[4];

    // Prologue: stage first tile.
    load_tile_regs<BLAYOUT>(A, Bp, K, N, tile_m, tile_n, 0, tid, sa, sb);
    store_tile_lds<BLAYOUT>(lds_a[0], lds_bt[0], tid, sa, sb);
    __syncthreads();

    int buf = 0;
    for (int k0 = 0; k0 < K; k0 += TILE_K) {
        const bool more = (k0 + TILE_K) < K;
        // Issue next tile's global loads before compute: latency hides under WMMA.
        if (more)
            load_tile_regs<BLAYOUT>(A, Bp, K, N, tile_m, tile_n, k0 + TILE_K, tid, sa, sb);

        v16h af[4], bf[2];
        #pragma unroll
        for (int i = 0; i < 4; ++i)
            af[i] = load_a_frag(&lds_a[buf][wm * 64 + i * 16 + lr][0], lane);
        #pragma unroll
        for (int j = 0; j < 2; ++j)
            bf[j] = *(const v16h*)&lds_bt[buf][wn * 32 + j * 16 + lr][kh];

        #pragma unroll
        for (int i = 0; i < 4; ++i)
            #pragma unroll
            for (int j = 0; j < 2; ++j)
                acc[i][j] = __builtin_amdgcn_wmma_f32_16x16x32_f16(
                    false, af[i], false, bf[j], (short)0, acc[i][j], false, false);

        if (more)
            store_tile_lds<BLAYOUT>(lds_a[buf ^ 1], lds_bt[buf ^ 1], tid, sa, sb);
        __syncthreads();
        buf ^= 1;
    }

    // ---- epilogue + store ----
    // C/D fragment: VGPR v, lanes 0-15: M=v, N=lane; lanes 16-31: M=v+8, N=lane-16
    const int row_half = (lane >> 4) * 8;
    #pragma unroll
    for (int i = 0; i < 4; ++i) {
        const int row0 = tile_m + wm * 64 + i * 16 + row_half;
        #pragma unroll
        for (int j = 0; j < 2; ++j) {
            const int col = tile_n + wn * 32 + j * 16 + lr;
            #pragma unroll
            for (int v = 0; v < 8; ++v) {
                const int row = row0 + v;
                float x = acc[i][j][v];
                if (EPI == 0) {
                    x += bias[col];
                } else if (EPI == 1) {
                    float mval = (float)maskp[(long long)b * sMask + (long long)col * ldm + row];
                    x = x * scale + mval * (-1e9f);
                } else if (EPI == 2) {
                    x += bias[col];
                    x = 0.5f * x * (1.0f + erff(x * 0.70710678118654752f));
                }
                C[(long long)row * N + col] = x;
            }
        }
    }
}

// Row softmax, in place; one block per row.
__global__ __launch_bounds__(256)
void softmax_rows(float* __restrict__ data, int n)
{
    __shared__ float red[256];
    float* row = data + (long long)blockIdx.x * n;
    const int tid = threadIdx.x;

    float m = -3.4e38f;
    for (int i = tid; i < n; i += 256) m = fmaxf(m, row[i]);
    red[tid] = m; __syncthreads();
    for (int s = 128; s > 0; s >>= 1) { if (tid < s) red[tid] = fmaxf(red[tid], red[tid + s]); __syncthreads(); }
    m = red[0]; __syncthreads();

    float sum = 0.0f;
    for (int i = tid; i < n; i += 256) { float e = expf(row[i] - m); row[i] = e; sum += e; }
    red[tid] = sum; __syncthreads();
    for (int s = 128; s > 0; s >>= 1) { if (tid < s) red[tid] += red[tid + s]; __syncthreads(); }
    const float inv = 1.0f / red[0];
    __syncthreads();
    for (int i = tid; i < n; i += 256) row[i] *= inv;
}

// Row LayerNorm; one block per row.
__global__ __launch_bounds__(256)
void layernorm_rows(const float* __restrict__ in, float* __restrict__ out,
                    const float* __restrict__ gamma, const float* __restrict__ beta, int n)
{
    __shared__ float red[256];
    const float* row = in + (long long)blockIdx.x * n;
    float* orow = out + (long long)blockIdx.x * n;
    const int tid = threadIdx.x;

    float s = 0.0f;
    for (int i = tid; i < n; i += 256) s += row[i];
    red[tid] = s; __syncthreads();
    for (int st = 128; st > 0; st >>= 1) { if (tid < st) red[tid] += red[tid + st]; __syncthreads(); }
    const float mu = red[0] / (float)n;
    __syncthreads();

    float v = 0.0f;
    for (int i = tid; i < n; i += 256) { float d = row[i] - mu; v += d * d; }
    red[tid] = v; __syncthreads();
    for (int st = 128; st > 0; st >>= 1) { if (tid < st) red[tid] += red[tid + st]; __syncthreads(); }
    const float inv = rsqrtf(red[0] / (float)n + 1e-5f);
    __syncthreads();

    for (int i = tid; i < n; i += 256)
        orow[i] = (row[i] - mu) * inv * gamma[i] + beta[i];
}

extern "C" void kernel_launch(void* const* d_in, const int* in_sizes, int n_in,
                              void* d_out, int out_size, void* d_ws, size_t ws_size,
                              hipStream_t stream)
{
    (void)in_sizes; (void)n_in; (void)out_size; (void)ws_size;
    const int Bd = 8, L = 2048, D = 1024, Ad = 1024, VD = 1024, H = 1024;

    const float* Q    = (const float*)d_in[0];
    const float* Kin  = (const float*)d_in[1];
    const float* V    = (const float*)d_in[2];
    const int*   mask = (const int*)  d_in[3];
    const float* Wq   = (const float*)d_in[4];
    const float* bq   = (const float*)d_in[5];
    const float* Wk   = (const float*)d_in[6];
    const float* bk   = (const float*)d_in[7];
    const float* Wv   = (const float*)d_in[8];
    const float* bv   = (const float*)d_in[9];
    const float* Wo   = (const float*)d_in[10];
    const float* bo   = (const float*)d_in[11];
    const float* gamma= (const float*)d_in[12];
    const float* beta = (const float*)d_in[13];

    float* ctx_out  = (float*)d_out;                       // [B,L,H]
    float* attn_out = ctx_out + (long long)Bd * L * H;     // [B,Lk,Lq]

    const long long S = (long long)Bd * L * Ad;            // 16M floats per region
    float* qp   = (float*)d_ws;      // q projection  [B,L,A]
    float* kp   = qp + S;            // k projection  [B,L,A]
    float* wvp  = kp + S;            // v projection  [B,L,VD]
    float* ctx  = wvp + S;           // attn @ wv     [B,L,VD]
    float* hbuf = qp;                // pre-LN buffer (reuses qp; dead after scores)

    const dim3 blk(GTHREADS);

    // ---- projections: [B*L,D] @ [D,N] + bias ----
    {
        dim3 grd(Ad / TILE_N, (Bd * L) / TILE_M, 1);
        wmma_gemm<0, 0><<<grd, blk, 0, stream>>>(Q,   0, Wq, 0, qp,  0, bq, nullptr, 0, 0, Bd * L, Ad, D, 1.0f);
        wmma_gemm<0, 0><<<grd, blk, 0, stream>>>(Kin, 0, Wk, 0, kp,  0, bk, nullptr, 0, 0, Bd * L, Ad, D, 1.0f);
        wmma_gemm<0, 0><<<grd, blk, 0, stream>>>(V,   0, Wv, 0, wvp, 0, bv, nullptr, 0, 0, Bd * L, VD, D, 1.0f);
    }
    // ---- scores, already transposed: S[b,k,q] = (kp @ qp^T)/sqrt(A) + mask[b,q,k]*(-1e9) ----
    {
        dim3 grd(L / TILE_N, L / TILE_M, Bd);
        wmma_gemm<1, 1><<<grd, blk, 0, stream>>>(kp, (long long)L * Ad, qp, (long long)L * Ad,
                                                 attn_out, (long long)L * L,
                                                 nullptr, mask, (long long)L * L, L,
                                                 L, L, Ad, 0.03125f);
    }
    // ---- softmax over q axis (contiguous rows), in place in d_out ----
    softmax_rows<<<dim3(Bd * L), blk, 0, stream>>>(attn_out, L);
    // ---- context[b,k,v] = attn[b,k,:] @ wv[b,:,v] ----
    {
        dim3 grd(VD / TILE_N, L / TILE_M, Bd);
        wmma_gemm<0, 3><<<grd, blk, 0, stream>>>(attn_out, (long long)L * L, wvp, (long long)L * VD,
                                                 ctx, (long long)L * VD,
                                                 nullptr, nullptr, 0, 0, L, VD, L, 1.0f);
    }
    // ---- output projection + GELU ----
    {
        dim3 grd(H / TILE_N, (Bd * L) / TILE_M, 1);
        wmma_gemm<0, 2><<<grd, blk, 0, stream>>>(ctx, 0, Wo, 0, hbuf, 0, bo, nullptr, 0, 0, Bd * L, H, VD, 1.0f);
    }
    // ---- LayerNorm -> final output ----
    layernorm_rows<<<dim3(Bd * L), blk, 0, stream>>>(hbuf, ctx_out, gamma, beta, H);
}
